// NeuronMiniMaxM2DecoderLayer_63127429316824
// MI455X (gfx1250) — matrix operand order repaired and checked by reading, not compiled, not run on hardware
//
#include <hip/hip_runtime.h>
#include <cstdint>

// ---------------------------------------------------------------------------
// Types for CDNA5 WMMA (gfx1250, wave32)
// ---------------------------------------------------------------------------
typedef __attribute__((ext_vector_type(16))) __bf16 v16bf;
typedef __attribute__((ext_vector_type(8)))  float  v8f;

#define B_  1
#define S_  2048
#define H_  1024
#define NH_ 8
#define HD_ 128
#define NKV_ 2
#define RD_ 64
#define E_  8
#define TOPK_ 2
#define I_  1024

__device__ __forceinline__ v8f wmma_bf16(v16bf a, v16bf b, v8f c) {
  return __builtin_amdgcn_wmma_f32_16x16x32_bf16(false, a, false, b, (short)0, c,
                                                 false, false);
}

// A-fragment (16x32, 16-bit): lane m = lane&15, K-half kb = (lane>=16)?8:0.
// elements i0..7 -> K = kb..kb+7 ; i8..15 -> K = kb+16..kb+23
__device__ __forceinline__ v16bf load_frag_a(const __bf16* tile, int row0, int k0,
                                             int stride) {
  int lane = threadIdx.x & 31;
  int m = lane & 15;
  int kb = (lane >> 4) * 8;
  const __bf16* p = tile + (row0 + m) * stride + k0 + kb;
  union { uint4 u[2]; v16bf v; } un;
  un.u[0] = *(const uint4*)p;
  un.u[1] = *(const uint4*)(p + 16);
  return un.v;
}

// B-fragment (32x16, 16-bit) from an n-major (transposed) LDS tile:
// lane n = lane&15, kb = (lane>=16)?8:0, i0..7 -> K = kb..kb+7 ; i8..15 -> K = kb+16..kb+23
__device__ __forceinline__ v16bf load_frag_b(const __bf16* tileT, int n0, int k0,
                                             int stride) {
  int lane = threadIdx.x & 31;
  int n = lane & 15;
  int kb = (lane >> 4) * 8;
  const __bf16* p = tileT + (n0 + n) * stride + k0 + kb;
  union { uint4 u[2]; v16bf v; } un;
  un.u[0] = *(const uint4*)p;
  un.u[1] = *(const uint4*)(p + 16);
  return un.v;
}

// ---------------------------------------------------------------------------
// RMSNorm: f32 [rows x width] -> bf16, blockDim = 256
// ---------------------------------------------------------------------------
__global__ void rmsnorm_kernel(const float* __restrict__ x,
                               const float* __restrict__ w,
                               __bf16* __restrict__ out, int width) {
  int t = blockIdx.x;
  const float* xr = x + (size_t)t * width;
  float ss = 0.f;
  for (int i = threadIdx.x; i < width; i += blockDim.x) {
    float v = xr[i];
    ss += v * v;
  }
  for (int d = 1; d < 32; d <<= 1) ss += __shfl_xor(ss, d, 32);
  __shared__ float red[8];
  int wid = threadIdx.x >> 5, lane = threadIdx.x & 31;
  if (lane == 0) red[wid] = ss;
  __syncthreads();
  int nw = blockDim.x >> 5;
  float tot = 0.f;
  for (int i = 0; i < nw; ++i) tot += red[i];
  float r = rsqrtf(tot / (float)width + 1e-6f);
  __bf16* orow = out + (size_t)t * width;
  for (int i = threadIdx.x; i < width; i += blockDim.x)
    orow[i] = (__bf16)(xr[i] * r * w[i]);
}

// ---------------------------------------------------------------------------
// Fused RMSNorm + partial RoPE (RD=64 of each 128-wide head), blockDim = 128
// ---------------------------------------------------------------------------
__global__ void rope_norm_kernel(const float* __restrict__ raw,
                                 const float* __restrict__ w,
                                 __bf16* __restrict__ out, int width) {
  int t = blockIdx.x;
  const float* xr = raw + (size_t)t * width;
  float ss = 0.f;
  for (int i = threadIdx.x; i < width; i += blockDim.x) {
    float v = xr[i];
    ss += v * v;
  }
  for (int d = 1; d < 32; d <<= 1) ss += __shfl_xor(ss, d, 32);
  __shared__ float red[8];
  int wid = threadIdx.x >> 5, lane = threadIdx.x & 31;
  if (lane == 0) red[wid] = ss;
  __syncthreads();
  int nw = blockDim.x >> 5;
  float tot = 0.f;
  for (int i = 0; i < nw; ++i) tot += red[i];
  float r = rsqrtf(tot / (float)width + 1e-6f);
  __bf16* orow = out + (size_t)t * width;
  for (int i = threadIdx.x; i < width; i += blockDim.x) {
    int d = i & 127;
    float val;
    if (d < 64) {
      int j = d & 31;
      // inv_freq_j = THETA^(-j/32), THETA = 1e6 ; ln(1e6)/32 = 0.4317347
      float freq = __expf(-0.43173470f * (float)j);
      float ang = (float)t * freq;
      float s, c;
      __sincosf(ang, &s, &c);
      float xa = xr[i] * r * w[i];
      int ip = (d < 32) ? (i + 32) : (i - 32);
      float xb = xr[ip] * r * w[ip];
      val = (d < 32) ? (xa * c - xb * s) : (xa * c + xb * s);
    } else {
      val = xr[i] * r * w[i];
    }
    orow[i] = (__bf16)val;
  }
}

// ---------------------------------------------------------------------------
// Generic WMMA GEMM: C[M,N] = A_bf16[M,K] @ B_f32[K,N]   (B -> bf16 in LDS)
// blockDim = 256 (8 waves): block tile 128x128, wave tile 32x64, K-step 32.
// 8 WMMAs per wave per K-step for 6 fragment loads (12 ds_load_b128).
// Supports: expert batching (blockIdx.z + cnt/base), row gather, epilogues:
//   0: f32 store        1: bf16 store
//   2: f32 store + residual add       3: bf16 store of silu(gate)*C
// ---------------------------------------------------------------------------
__global__ void gemm_wmma(const __bf16* __restrict__ A, int lda,
                          const float* __restrict__ B, int ldb, long long strideB,
                          const int* __restrict__ gather,
                          const int* __restrict__ cntPtr,
                          const int* __restrict__ basePtr,
                          int M, int N, int K,
                          float* __restrict__ outF32,
                          __bf16* __restrict__ outBf16,
                          const float* __restrict__ gateBuf,
                          const float* __restrict__ resid,
                          int ldc, int epi) {
  int e = blockIdx.z;
  int rows = M, base = 0;
  if (cntPtr) { rows = cntPtr[e]; base = basePtr[e]; }
  int m0 = blockIdx.y * 128;
  if (m0 >= rows) return;
  int n0 = blockIdx.x * 128;
  const float* Bx = B + (long long)e * strideB;

  __shared__ __align__(16) __bf16 Atile[128 * 40];
  __shared__ __align__(16) __bf16 Btile[128 * 40];  // n-major (transposed)

  int tid = threadIdx.x;
  int lane = tid & 31, wid = tid >> 5;
  int wm = wid & 3, wn = wid >> 2;
  int half = lane >> 4, nloc = lane & 15;

  // A loader mapping: 128 rows x 32 cols, 2 threads/row (16 bf16 each)
  int arow = tid >> 1;
  int ahalf = (tid & 1) * 16;
  int lr_load = m0 + arow;
  int clr = lr_load < rows - 1 ? lr_load : rows - 1;
  int rowIdx;
  if (gather) rowIdx = gather[base + clr] & (S_ - 1);
  else        rowIdx = base + clr;
  // B loader mapping: 32 rows x 128 cols, 8 threads/row (16 f32 each)
  int bkr = tid >> 3;
  int bnc = (tid & 7) * 16;

  v8f acc[2][4] = {};

  for (int k0 = 0; k0 < K; k0 += 32) {
    __syncthreads();
    {
      const __bf16* pa = A + (size_t)rowIdx * lda + k0 + ahalf;
      uint4 u0 = *(const uint4*)pa;
      uint4 u1 = *(const uint4*)(pa + 8);
      *(uint4*)&Atile[arow * 40 + ahalf] = u0;
      *(uint4*)&Atile[arow * 40 + ahalf + 8] = u1;
    }
    {
      const float* pb = Bx + (size_t)(k0 + bkr) * ldb + n0 + bnc;
      float4 f0 = *(const float4*)pb;
      float4 f1 = *(const float4*)(pb + 4);
      float4 f2 = *(const float4*)(pb + 8);
      float4 f3 = *(const float4*)(pb + 12);
      __bf16 tmp[16] = {(__bf16)f0.x, (__bf16)f0.y, (__bf16)f0.z, (__bf16)f0.w,
                        (__bf16)f1.x, (__bf16)f1.y, (__bf16)f1.z, (__bf16)f1.w,
                        (__bf16)f2.x, (__bf16)f2.y, (__bf16)f2.z, (__bf16)f2.w,
                        (__bf16)f3.x, (__bf16)f3.y, (__bf16)f3.z, (__bf16)f3.w};
#pragma unroll
      for (int j = 0; j < 16; ++j) Btile[(bnc + j) * 40 + bkr] = tmp[j];
    }
    // prefetch next K-slab into L2/WGP while this tile computes
    if (k0 + 32 < K) {
      __builtin_prefetch(A + (size_t)rowIdx * lda + k0 + 32 + ahalf, 0, 1);
      __builtin_prefetch(Bx + (size_t)(k0 + 32 + bkr) * ldb + n0 + bnc, 0, 1);
    }
    __syncthreads();

    v16bf a0 = load_frag_a(Atile, wm * 32, 0, 40);
    v16bf a1 = load_frag_a(Atile, wm * 32 + 16, 0, 40);
#pragma unroll
    for (int nt = 0; nt < 4; ++nt) {
      v16bf b = load_frag_b(Btile, wn * 64 + nt * 16, 0, 40);
      acc[0][nt] = wmma_bf16(a0, b, acc[0][nt]);
      acc[1][nt] = wmma_bf16(a1, b, acc[1][nt]);
    }
  }

#pragma unroll
  for (int mt = 0; mt < 2; ++mt)
#pragma unroll
    for (int nt = 0; nt < 4; ++nt)
#pragma unroll
      for (int r = 0; r < 8; ++r) {
        int lr = m0 + wm * 32 + mt * 16 + half * 8 + r;
        if (lr >= rows) continue;
        size_t gr = (size_t)(base + lr);
        int col = n0 + wn * 64 + nt * 16 + nloc;
        float a = acc[mt][nt][r];
        size_t idx = gr * (size_t)ldc + col;
        if (epi == 0) {
          outF32[idx] = a;
        } else if (epi == 1) {
          outBf16[idx] = (__bf16)a;
        } else if (epi == 2) {
          outF32[idx] = resid[idx] + a;
        } else {
          float g = gateBuf[idx];
          float sg = g / (1.f + __expf(-g));
          outBf16[idx] = (__bf16)(sg * a);
        }
      }
}

// ---------------------------------------------------------------------------
// Flash attention (causal, GQA): grid (S/64, NH), blockDim = 128 (4 waves)
// Each wave owns 16 q rows; KV tiles of 32 rows staged in LDS.
// ---------------------------------------------------------------------------
__global__ void flash_attn(const __bf16* __restrict__ qb,
                           const __bf16* __restrict__ kb,
                           const __bf16* __restrict__ vb,
                           __bf16* __restrict__ ob) {
  const int q0 = blockIdx.x * 64;
  const int head = blockIdx.y;
  const int kvh = head >> 2;  // NH / NKV = 4
  const int tid = threadIdx.x, lane = tid & 31, wid = tid >> 5;
  const int half = lane >> 4, nloc = lane & 15;

  __shared__ __align__(16) __bf16 Qs[64 * 136];
  __shared__ __align__(16) __bf16 Ks[32 * 136];
  __shared__ __align__(16) __bf16 Vt[128 * 40];   // n-major (hd x kvpos)
  __shared__ __align__(16) __bf16 Ps[4 * 16 * 40];

  {
    int r = tid >> 1, ch = (tid & 1) * 64;
    const __bf16* pq = qb + (size_t)(q0 + r) * (NH_ * HD_) + head * HD_ + ch;
    uint4* dst = (uint4*)&Qs[r * 136 + ch];
    const uint4* src = (const uint4*)pq;
    dst[0] = src[0]; dst[1] = src[1]; dst[2] = src[2]; dst[3] = src[3];
  }

  v8f o[8];
  v8f z = {};
#pragma unroll
  for (int i = 0; i < 8; ++i) o[i] = z;
  float mrow[8], lrow[8];
#pragma unroll
  for (int r = 0; r < 8; ++r) { mrow[r] = -1e30f; lrow[r] = 0.f; }
  const float scale = 0.08838834764831845f;  // 1/sqrt(128)

  const int ktEnd = (q0 + 63) / 32;
  for (int kt = 0; kt <= ktEnd; ++kt) {
    __syncthreads();
    {
      int r = tid >> 2, cs = (tid & 3) * 32;
      const __bf16* pk = kb + (size_t)(kt * 32 + r) * (NKV_ * HD_) + kvh * HD_ + cs;
      uint4* dst = (uint4*)&Ks[r * 136 + cs];
      const uint4* src = (const uint4*)pk;
      dst[0] = src[0]; dst[1] = src[1]; dst[2] = src[2]; dst[3] = src[3];
      const __bf16* pv = vb + (size_t)(kt * 32 + r) * (NKV_ * HD_) + kvh * HD_ + cs;
      union { uint4 u[4]; __bf16 e[32]; } tv;
      const uint4* vs = (const uint4*)pv;
      tv.u[0] = vs[0]; tv.u[1] = vs[1]; tv.u[2] = vs[2]; tv.u[3] = vs[3];
#pragma unroll
      for (int j = 0; j < 32; ++j) Vt[(cs + j) * 40 + r] = tv.e[j];
    }
    __syncthreads();

    // S = Q @ K^T  (Ks row-major over kv-pos doubles as transposed-B tile)
    v8f s0 = {}, s1 = {};
#pragma unroll
    for (int kk = 0; kk < 4; ++kk) {
      v16bf a = load_frag_a(Qs, wid * 16, kk * 32, 136);
      v16bf b0 = load_frag_b(Ks, 0, kk * 32, 136);
      v16bf b1 = load_frag_b(Ks, 16, kk * 32, 136);
      s0 = wmma_bf16(a, b0, s0);
      s1 = wmma_bf16(a, b1, s1);
    }

    // online softmax
    __bf16* Pw = &Ps[wid * 16 * 40];
#pragma unroll
    for (int r = 0; r < 8; ++r) {
      int row = q0 + wid * 16 + half * 8 + r;
      int c0 = kt * 32 + nloc, c1 = c0 + 16;
      float x0 = (c0 <= row) ? s0[r] * scale : -1e30f;
      float x1 = (c1 <= row) ? s1[r] * scale : -1e30f;
      float m = fmaxf(x0, x1);
      for (int d = 1; d < 16; d <<= 1) m = fmaxf(m, __shfl_xor(m, d, 32));
      float mn = fmaxf(mrow[r], m);
      float al = __expf(mrow[r] - mn);
      float p0 = __expf(x0 - mn);
      float p1 = __expf(x1 - mn);
      float rs = p0 + p1;
      for (int d = 1; d < 16; d <<= 1) rs += __shfl_xor(rs, d, 32);
      lrow[r] = lrow[r] * al + rs;
      mrow[r] = mn;
#pragma unroll
      for (int n2 = 0; n2 < 8; ++n2) o[n2][r] *= al;
      Pw[(half * 8 + r) * 40 + nloc] = (__bf16)p0;
      Pw[(half * 8 + r) * 40 + nloc + 16] = (__bf16)p1;
    }
    // make the per-wave P spill visible before fragment reload
    asm volatile("s_wait_dscnt 0x0" ::: "memory");

    // O += P @ V
    v16bf pa = load_frag_a(Pw, 0, 0, 40);
#pragma unroll
    for (int n2 = 0; n2 < 8; ++n2) {
      v16bf b = load_frag_b(Vt, n2 * 16, 0, 40);
      o[n2] = wmma_bf16(pa, b, o[n2]);
    }
  }

#pragma unroll
  for (int n2 = 0; n2 < 8; ++n2)
#pragma unroll
    for (int r = 0; r < 8; ++r) {
      int row = q0 + wid * 16 + half * 8 + r;
      int col = head * HD_ + n2 * 16 + nloc;
      ob[(size_t)row * (NH_ * HD_) + col] = (__bf16)(o[n2][r] / lrow[r]);
    }
}

// ---------------------------------------------------------------------------
// Router: wave per token, sigmoid + bias top-2, normalized weights
// ---------------------------------------------------------------------------
__global__ void router_kernel(const __bf16* __restrict__ t2,
                              const float* __restrict__ Wr,
                              const float* __restrict__ bias,
                              int* __restrict__ topi, float* __restrict__ topw) {
  int t = blockIdx.x * 8 + (threadIdx.x >> 5);
  int lane = threadIdx.x & 31;
  float acc[E_];
#pragma unroll
  for (int e = 0; e < E_; ++e) acc[e] = 0.f;
  const __bf16* tr = t2 + (size_t)t * H_;
  for (int h = lane; h < H_; h += 32) {
    float tv = (float)tr[h];
    const float* wr = Wr + (size_t)h * E_;
#pragma unroll
    for (int e = 0; e < E_; ++e) acc[e] += tv * wr[e];
  }
#pragma unroll
  for (int e = 0; e < E_; ++e)
    for (int d = 1; d < 32; d <<= 1) acc[e] += __shfl_xor(acc[e], d, 32);
  if (lane == 0) {
    float sig[E_], sc[E_];
#pragma unroll
    for (int e = 0; e < E_; ++e) {
      sig[e] = 1.f / (1.f + __expf(-acc[e]));
      sc[e] = sig[e] + bias[e];
    }
    int i0 = 0;
    for (int e = 1; e < E_; ++e) if (sc[e] > sc[i0]) i0 = e;
    int i1 = -1;
    for (int e = 0; e < E_; ++e)
      if (e != i0 && (i1 < 0 || sc[e] > sc[i1])) i1 = e;
    float w0 = sig[i0], w1 = sig[i1];
    float s = w0 + w1;
    topi[t * 2] = i0;
    topi[t * 2 + 1] = i1;
    topw[t * 2] = w0 / s;
    topw[t * 2 + 1] = w1 / s;
  }
}

__global__ void moe_zero(int* cnt, int* fill) {
  if (threadIdx.x < E_) { cnt[threadIdx.x] = 0; fill[threadIdx.x] = 0; }
}
__global__ void moe_count(const int* __restrict__ topi, int* cnt) {
  int i = blockIdx.x * blockDim.x + threadIdx.x;
  if (i < TOPK_ * S_) atomicAdd(&cnt[topi[i]], 1);
}
__global__ void moe_prefix(const int* __restrict__ cnt, int* base) {
  if (threadIdx.x == 0) {
    int b = 0;
    for (int e = 0; e < E_; ++e) { base[e] = b; b += cnt[e]; }
  }
}
__global__ void moe_fill(const int* __restrict__ topi, const int* __restrict__ base,
                         int* fill, int* __restrict__ listTok, int* __restrict__ pos) {
  int i = blockIdx.x * blockDim.x + threadIdx.x;
  if (i < TOPK_ * S_) {
    int e = topi[i];
    int p = atomicAdd(&fill[e], 1);
    int idx = base[e] + p;
    listTok[idx] = i >> 1;
    pos[i] = idx;
  }
}

__global__ void combine_kernel(float* __restrict__ out, const float* __restrict__ ybuf,
                               const int* __restrict__ pos,
                               const float* __restrict__ topw) {
  int t = blockIdx.x;
  int p0 = pos[t * 2], p1 = pos[t * 2 + 1];
  float w0 = topw[t * 2], w1 = topw[t * 2 + 1];
  for (int h = threadIdx.x; h < H_; h += blockDim.x) {
    size_t i = (size_t)t * H_ + h;
    out[i] += w0 * ybuf[(size_t)p0 * H_ + h] + w1 * ybuf[(size_t)p1 * H_ + h];
  }
}

// ---------------------------------------------------------------------------
extern "C" void kernel_launch(void* const* d_in, const int* in_sizes, int n_in,
                              void* d_out, int out_size, void* d_ws, size_t ws_size,
                              hipStream_t stream) {
  (void)in_sizes; (void)n_in; (void)out_size; (void)ws_size;

  const float* x       = (const float*)d_in[0];
  const float* in_ln   = (const float*)d_in[1];
  const float* post_ln = (const float*)d_in[2];
  const float* qn_w    = (const float*)d_in[3];
  const float* kn_w    = (const float*)d_in[4];
  const float* Wq      = (const float*)d_in[5];
  const float* Wk      = (const float*)d_in[6];
  const float* Wv      = (const float*)d_in[7];
  const float* Wo      = (const float*)d_in[8];
  const float* Wr      = (const float*)d_in[9];
  const float* rbias   = (const float*)d_in[10];
  const float* Wg      = (const float*)d_in[11];
  const float* Wu      = (const float*)d_in[12];
  const float* Wd      = (const float*)d_in[13];
  float* out = (float*)d_out;

  char* wsp = (char*)d_ws;
  auto alloc = [&](size_t bytes) -> char* {
    char* p = wsp;
    wsp += (bytes + 255) & ~(size_t)255;
    return p;
  };
  __bf16* hb   = (__bf16*)alloc((size_t)S_ * H_ * 2);
  float*  qraw = (float*) alloc((size_t)S_ * NH_ * HD_ * 4);
  float*  kraw = (float*) alloc((size_t)S_ * NKV_ * HD_ * 4);
  __bf16* qbf  = (__bf16*)alloc((size_t)S_ * NH_ * HD_ * 2);
  __bf16* kbf  = (__bf16*)alloc((size_t)S_ * NKV_ * HD_ * 2);
  __bf16* vbf  = (__bf16*)alloc((size_t)S_ * NKV_ * HD_ * 2);
  __bf16* obf  = (__bf16*)alloc((size_t)S_ * NH_ * HD_ * 2);
  __bf16* h2b  = (__bf16*)alloc((size_t)S_ * H_ * 2);
  float*  gbuf = (float*) alloc((size_t)TOPK_ * S_ * I_ * 4);
  __bf16* abuf = (__bf16*)alloc((size_t)TOPK_ * S_ * I_ * 2);
  float*  ybuf = (float*) alloc((size_t)TOPK_ * S_ * H_ * 4);
  int*    topi = (int*)   alloc((size_t)TOPK_ * S_ * 4);
  float*  topw = (float*) alloc((size_t)TOPK_ * S_ * 4);
  int*    pos  = (int*)   alloc((size_t)TOPK_ * S_ * 4);
  int*    lstT = (int*)   alloc((size_t)TOPK_ * S_ * 4);
  int*    cnt  = (int*)   alloc(64);
  int*    base = (int*)   alloc(64);
  int*    fill = (int*)   alloc(64);

  // 1) input RMSNorm
  rmsnorm_kernel<<<S_, 256, 0, stream>>>(x, in_ln, hb, H_);

  // 2) QKV projections
  gemm_wmma<<<dim3(8, 16, 1), 256, 0, stream>>>(hb, H_, Wq, NH_ * HD_, 0, nullptr,
      nullptr, nullptr, S_, NH_ * HD_, H_, qraw, nullptr, nullptr, nullptr,
      NH_ * HD_, 0);
  gemm_wmma<<<dim3(2, 16, 1), 256, 0, stream>>>(hb, H_, Wk, NKV_ * HD_, 0, nullptr,
      nullptr, nullptr, S_, NKV_ * HD_, H_, kraw, nullptr, nullptr, nullptr,
      NKV_ * HD_, 0);
  gemm_wmma<<<dim3(2, 16, 1), 256, 0, stream>>>(hb, H_, Wv, NKV_ * HD_, 0, nullptr,
      nullptr, nullptr, S_, NKV_ * HD_, H_, nullptr, vbf, nullptr, nullptr,
      NKV_ * HD_, 1);

  // 3) q/k norm + partial RoPE
  rope_norm_kernel<<<S_, 128, 0, stream>>>(qraw, qn_w, qbf, NH_ * HD_);
  rope_norm_kernel<<<S_, 128, 0, stream>>>(kraw, kn_w, kbf, NKV_ * HD_);

  // 4) causal GQA flash attention
  flash_attn<<<dim3(S_ / 64, NH_, 1), 128, 0, stream>>>(qbf, kbf, vbf, obf);

  // 5) O projection + residual -> d_out (= x1)
  gemm_wmma<<<dim3(8, 16, 1), 256, 0, stream>>>(obf, NH_ * HD_, Wo, H_, 0, nullptr,
      nullptr, nullptr, S_, H_, NH_ * HD_, out, nullptr, nullptr, x, H_, 2);

  // 6) post RMSNorm + router + dispatch
  rmsnorm_kernel<<<S_, 256, 0, stream>>>(out, post_ln, h2b, H_);
  router_kernel<<<S_ / 8, 256, 0, stream>>>(h2b, Wr, rbias, topi, topw);
  moe_zero<<<1, 32, 0, stream>>>(cnt, fill);
  moe_count<<<16, 256, 0, stream>>>(topi, cnt);
  moe_prefix<<<1, 32, 0, stream>>>(cnt, base);
  moe_fill<<<16, 256, 0, stream>>>(topi, base, fill, lstT, pos);

  // 7) sparse MoE (top-2): grouped GEMMs per expert
  gemm_wmma<<<dim3(8, 16, E_), 256, 0, stream>>>(h2b, H_, Wg, I_,
      (long long)H_ * I_, lstT, cnt, base, S_, I_, H_, gbuf, nullptr, nullptr,
      nullptr, I_, 0);
  gemm_wmma<<<dim3(8, 16, E_), 256, 0, stream>>>(h2b, H_, Wu, I_,
      (long long)H_ * I_, lstT, cnt, base, S_, I_, H_, nullptr, abuf, gbuf,
      nullptr, I_, 3);
  gemm_wmma<<<dim3(8, 16, E_), 256, 0, stream>>>(abuf, I_, Wd, H_,
      (long long)I_ * H_, nullptr, cnt, base, S_, H_, I_, ybuf, nullptr, nullptr,
      nullptr, H_, 0);

  // 8) weighted combine into residual stream
  combine_kernel<<<S_, 256, 0, stream>>>(out, ybuf, pos, topw);
}
